// QuantLinear_10419590660824
// MI455X (gfx1250) — compile-verified, matrix operand order checked by reading, not run on hardware
//
#include <hip/hip_runtime.h>
#include <stdint.h>

typedef __attribute__((ext_vector_type(16))) __bf16 v16bf;
typedef __attribute__((ext_vector_type(8)))  float  v8f;
typedef __attribute__((ext_vector_type(4)))  int    v4i;

#define TILE_M   128
#define TILE_N   128
#define TILE_K   128   // == GPTQ group size
#define A_STRIDE 136   // bf16 elems per LDS row (16B aligned, bank-skewed)
#define B_STRIDE 136

// ---------- CDNA5 async-to-LDS path (probe via __has_builtin) ----------
#if defined(__has_builtin)
#  if __has_builtin(__builtin_amdgcn_global_load_async_to_lds_b128)
#    define USE_ASYNC_LDS 1
#  endif
#endif
#ifndef USE_ASYNC_LDS
#  define USE_ASYNC_LDS 0
#endif

#if USE_ASYNC_LDS
__device__ __forceinline__ void cp_async_b128(void* lds, const void* g) {
  __builtin_amdgcn_global_load_async_to_lds_b128(
      (__attribute__((address_space(1))) v4i*)(g),
      (__attribute__((address_space(3))) v4i*)(lds), 0, 0);
}
#endif

__device__ __forceinline__ void wait_async_all() {
#if USE_ASYNC_LDS
#  if __has_builtin(__builtin_amdgcn_s_wait_asynccnt)
  __builtin_amdgcn_s_wait_asynccnt(0);
#  else
  asm volatile("s_wait_asynccnt 0" ::: "memory");
#  endif
#endif
}

// ---------- helpers ----------
__device__ __forceinline__ uint32_t pack_bf16(float a, float b) {
  uint32_t ua = __float_as_uint(a);
  uint32_t ub = __float_as_uint(b);
  ua = (ua + 0x7FFFu + ((ua >> 16) & 1u)) >> 16;
  ub = (ub + 0x7FFFu + ((ub >> 16) & 1u)) >> 16;
  return ua | (ub << 16);
}

// CDNA5 16-bit operand layout: lane L (row = base + L%16) holds
// K = {(L/16)*8 .. +7} in elems 0..7 and K = {16+(L/16)*8 .. +7} in 8..15.
__device__ __forceinline__ v16bf load_frag(const uint16_t* __restrict__ lds,
                                           int row, int stride, int kbase,
                                           int hlf) {
  union { uint4 q[2]; v16bf v; } u;
  const uint16_t* p = lds + row * stride + kbase + hlf * 8;
  u.q[0] = *(const uint4*)(p);
  u.q[1] = *(const uint4*)(p + 16);
  return u.v;
}

// ---------- prep kernel 1: x fp32 -> bf16 (once) ----------
__global__ __launch_bounds__(256) void convert_x_kernel(
    const float* __restrict__ x, uint16_t* __restrict__ xbf) {
  const size_t i = (size_t)blockIdx.x * blockDim.x + threadIdx.x;  // per 8 elems
  const float4* p = (const float4*)x + i * 2;
  float4 a = p[0];
  float4 b = p[1];
  uint4 v;
  v.x = pack_bf16(a.x, a.y);
  v.y = pack_bf16(a.z, a.w);
  v.z = pack_bf16(b.x, b.y);
  v.w = pack_bf16(b.z, b.w);
  ((uint4*)xbf)[i] = v;
}

// ---------- prep kernel 2: int4 -> bf16 W in [n][k] layout (once) ----------
__global__ __launch_bounds__(256) void dequant_w_kernel(
    const int* __restrict__ qweight, const int* __restrict__ qzeros,
    const float* __restrict__ scales, uint16_t* __restrict__ wbf,
    int N, int K) {
  const int n  = blockIdx.x * 256 + threadIdx.x;
  const int k0 = blockIdx.y * 64;
  const int g  = k0 >> 7;
  const uint32_t qz = (uint32_t)qzeros[g * (N >> 3) + (n >> 3)];
  const int z = (int)((((qz >> ((n & 7) * 4)) & 0xFu) + 1u) & 0xFu);
  const float s = scales[(size_t)g * N + n];
  const int* qcol = qweight + (size_t)(k0 >> 3) * N + n;
  uint16_t* wrow = wbf + (size_t)n * K + k0;  // 128B contiguous per thread
#pragma unroll
  for (int r = 0; r < 8; ++r) {
    const uint32_t q = (uint32_t)qcol[(size_t)r * N];
    uint4 v;
    uint32_t w[4];
#pragma unroll
    for (int p = 0; p < 4; ++p) {
      const int w0 = (int)((q >> (8 * p)) & 0xFu) - z;
      const int w1 = (int)((q >> (8 * p + 4)) & 0xFu) - z;
      w[p] = pack_bf16((float)w0 * s, (float)w1 * s);
    }
    v.x = w[0]; v.y = w[1]; v.z = w[2]; v.w = w[3];
    *(uint4*)(wrow + r * 8) = v;
  }
}

// ---------- stage one 128x128 bf16 tile pair (A and B), 4 waves ----------
__device__ __forceinline__ void stage_pair4(const uint16_t* __restrict__ xa,
                                            const uint16_t* __restrict__ wb,
                                            uint16_t* __restrict__ Ab,
                                            uint16_t* __restrict__ Bb,
                                            int wave, int lane, int K) {
  const int rowoff = lane >> 4;          // 0/1
  const int c16 = (lane & 15) * 8;       // bf16 chunk offset (16B)
#pragma unroll
  for (int rr = 0; rr < 16; ++rr) {
    const int r = rr * 8 + wave * 2 + rowoff;   // 4 waves x 2 rows x 16 iters
    const size_t go = (size_t)r * K + c16;
#if USE_ASYNC_LDS
    cp_async_b128(&Ab[r * A_STRIDE + c16], xa + go);
    cp_async_b128(&Bb[r * B_STRIDE + c16], wb + go);
#else
    *(uint4*)&Ab[r * A_STRIDE + c16] = *(const uint4*)(xa + go);
    *(uint4*)&Bb[r * B_STRIDE + c16] = *(const uint4*)(wb + go);
#endif
  }
}

// ---------- main bf16 GEMM: 128 threads, 64x64 wave tile, async DB ----------
__global__ __launch_bounds__(128) void bf16_gemm_kernel(
    const uint16_t* __restrict__ xbf, const uint16_t* __restrict__ wbf,
    const float* __restrict__ bias, float* __restrict__ out,
    int M, int N, int K) {
  __shared__ __align__(16) uint16_t Alds[2][TILE_M * A_STRIDE];
  __shared__ __align__(16) uint16_t Blds[2][TILE_N * B_STRIDE];

  const int tid  = threadIdx.x;
  const int wave = tid >> 5;    // 0..3
  const int lane = tid & 31;
  const int hlf  = lane >> 4;
  const int l16  = lane & 15;
  const int wm   = wave >> 1;   // 0..1 -> M offset wm*64
  const int wn   = wave & 1;    // 0..1 -> N offset wn*64
  const int n0   = blockIdx.x * TILE_N;
  const int m0   = blockIdx.y * TILE_M;

  const uint16_t* xa = xbf + (size_t)m0 * K;
  const uint16_t* wb = wbf + (size_t)n0 * K;

  v8f acc[4][4];
#pragma unroll
  for (int i = 0; i < 4; ++i)
#pragma unroll
    for (int j = 0; j < 4; ++j)
#pragma unroll
      for (int e = 0; e < 8; ++e) acc[i][j][e] = 0.0f;

  // preload tile 0
  stage_pair4(xa, wb, Alds[0], Blds[0], wave, lane, K);

  const int nIter = K / TILE_K;
  for (int it = 0; it < nIter; ++it) {
    wait_async_all();      // this wave's staged writes are in LDS
    __syncthreads();       // everyone's writes visible; prev reads done

    if (it + 1 < nIter)
      stage_pair4(xa + (size_t)(it + 1) * TILE_K,
                  wb + (size_t)(it + 1) * TILE_K, Alds[(it + 1) & 1],
                  Blds[(it + 1) & 1], wave, lane, K);

    const uint16_t* Ac = Alds[it & 1];
    const uint16_t* Bc = Blds[it & 1];
#pragma unroll
    for (int kk = 0; kk < 4; ++kk) {
      const int kb = kk * 32;
      v16bf afr[4], bfr[4];
#pragma unroll
      for (int mt = 0; mt < 4; ++mt)
        afr[mt] = load_frag(Ac, wm * 64 + mt * 16 + l16, A_STRIDE, kb, hlf);
#pragma unroll
      for (int nt = 0; nt < 4; ++nt)
        bfr[nt] = load_frag(Bc, wn * 64 + nt * 16 + l16, B_STRIDE, kb, hlf);
#pragma unroll
      for (int mt = 0; mt < 4; ++mt)
#pragma unroll
        for (int nt = 0; nt < 4; ++nt)
          acc[mt][nt] = __builtin_amdgcn_wmma_f32_16x16x32_bf16(
              false, afr[mt], false, bfr[nt], (short)0, acc[mt][nt], false,
              false);
    }
  }

#pragma unroll
  for (int nt = 0; nt < 4; ++nt) {
    const int col = n0 + wn * 64 + nt * 16 + l16;
    const float bv = bias[col];
#pragma unroll
    for (int mt = 0; mt < 4; ++mt) {
      const int rowb = m0 + wm * 64 + mt * 16 + 8 * hlf;
#pragma unroll
      for (int v = 0; v < 8; ++v)
        out[(size_t)(rowb + v) * N + col] = acc[mt][nt][v] + bv;
    }
  }
}

// ---------- fallback: fused dequant+GEMM (round-1 kernel) ----------
__global__ __launch_bounds__(256) void gptq_wmma_fused_kernel(
    const float* __restrict__ x, const int* __restrict__ qweight,
    const int* __restrict__ qzeros, const float* __restrict__ scales,
    const float* __restrict__ bias, float* __restrict__ out,
    int M, int N, int K) {
  __shared__ __align__(16) uint16_t Alds[TILE_M * A_STRIDE];
  __shared__ __align__(16) uint16_t Blds[TILE_N * B_STRIDE];

  const int tid  = threadIdx.x;
  const int wave = tid >> 5;
  const int lane = tid & 31;
  const int hlf  = lane >> 4;
  const int l16  = lane & 15;
  const int wm   = wave >> 1;
  const int wn   = wave & 1;
  const int n0   = blockIdx.x * TILE_N;
  const int m0   = blockIdx.y * TILE_M;
  const int zrow = N >> 3;

  v8f acc[2][4];
#pragma unroll
  for (int i = 0; i < 2; ++i)
#pragma unroll
    for (int j = 0; j < 4; ++j)
#pragma unroll
      for (int e = 0; e < 8; ++e) acc[i][j][e] = 0.0f;

  const int bn    = tid & 127;
  const int rbase = (tid >> 7) * 8;
  const int gn    = n0 + bn;
  const int zsh   = (gn & 7) * 4;

  for (int k0 = 0; k0 < K; k0 += TILE_K) {
#pragma unroll
    for (int rr = 0; rr < 16; ++rr) {
      const int r = wave + rr * 8;
      const float* px = x + (size_t)(m0 + r) * K + k0 + lane * 4;
      float4 f = *(const float4*)px;
      uint2 v;
      v.x = pack_bf16(f.x, f.y);
      v.y = pack_bf16(f.z, f.w);
      *(uint2*)&Alds[r * A_STRIDE + lane * 4] = v;
    }
    {
      const int g = k0 >> 7;
      const uint32_t qz = (uint32_t)qzeros[g * zrow + (gn >> 3)];
      const int z = (int)((((qz >> zsh) & 0xFu) + 1u) & 0xFu);
      const float s = scales[(size_t)g * N + gn];
      const int* qcol = qweight + (size_t)(k0 >> 3) * N + gn;
#pragma unroll
      for (int r = 0; r < 8; ++r) {
        const uint32_t q = (uint32_t)qcol[(size_t)(rbase + r) * N];
        uint4 v;
        uint32_t w[4];
#pragma unroll
        for (int p = 0; p < 4; ++p) {
          const int w0 = (int)((q >> (8 * p)) & 0xFu) - z;
          const int w1 = (int)((q >> (8 * p + 4)) & 0xFu) - z;
          w[p] = pack_bf16((float)w0 * s, (float)w1 * s);
        }
        v.x = w[0]; v.y = w[1]; v.z = w[2]; v.w = w[3];
        *(uint4*)&Blds[bn * B_STRIDE + (rbase + r) * 8] = v;
      }
    }
    __syncthreads();
#pragma unroll
    for (int kk = 0; kk < 4; ++kk) {
      const int kb = kk * 32;
      v16bf afr[2], bfr[4];
#pragma unroll
      for (int mt = 0; mt < 2; ++mt)
        afr[mt] = load_frag(Alds, wm * 32 + mt * 16 + l16, A_STRIDE, kb, hlf);
#pragma unroll
      for (int nt = 0; nt < 4; ++nt)
        bfr[nt] = load_frag(Blds, wn * 64 + nt * 16 + l16, B_STRIDE, kb, hlf);
#pragma unroll
      for (int mt = 0; mt < 2; ++mt)
#pragma unroll
        for (int nt = 0; nt < 4; ++nt)
          acc[mt][nt] = __builtin_amdgcn_wmma_f32_16x16x32_bf16(
              false, afr[mt], false, bfr[nt], (short)0, acc[mt][nt], false,
              false);
    }
    __syncthreads();
  }

#pragma unroll
  for (int nt = 0; nt < 4; ++nt) {
    const int col = n0 + wn * 64 + nt * 16 + l16;
    const float bv = bias[col];
#pragma unroll
    for (int mt = 0; mt < 2; ++mt) {
      const int rowb = m0 + wm * 32 + mt * 16 + 8 * hlf;
#pragma unroll
      for (int v = 0; v < 8; ++v)
        out[(size_t)(rowb + v) * N + col] = acc[mt][nt][v] + bv;
    }
  }
}

extern "C" void kernel_launch(void* const* d_in, const int* in_sizes, int n_in,
                              void* d_out, int out_size, void* d_ws,
                              size_t ws_size, hipStream_t stream) {
  const float* x       = (const float*)d_in[0];
  const int*   qweight = (const int*)d_in[1];
  const int*   qzeros  = (const int*)d_in[2];
  const float* scales  = (const float*)d_in[3];
  const float* bias    = (const float*)d_in[4];
  float*       out     = (float*)d_out;

  const int K = 4096;
  const int N = 11008;
  const int M = in_sizes[0] / K;  // 8192

  const size_t need = (size_t)M * K * 2 + (size_t)N * K * 2;
  dim3 grid(N / TILE_N, M / TILE_M);

  if (d_ws != nullptr && ws_size >= need) {
    uint16_t* xbf = (uint16_t*)d_ws;
    uint16_t* wbf = xbf + (size_t)M * K;
    const size_t xchunks = (size_t)M * K / 8;
    convert_x_kernel<<<(unsigned)(xchunks / 256), 256, 0, stream>>>(x, xbf);
    dequant_w_kernel<<<dim3(N / 256, K / 64), 256, 0, stream>>>(
        qweight, qzeros, scales, wbf, N, K);
    bf16_gemm_kernel<<<grid, dim3(128), 0, stream>>>(xbf, wbf, bias, out, M, N,
                                                     K);
  } else {
    gptq_wmma_fused_kernel<<<grid, dim3(256), 0, stream>>>(
        x, qweight, qzeros, scales, bias, out, M, N, K);
  }
}
